// Block_61564061221386
// MI455X (gfx1250) — matrix operand order repaired and checked by reading, not compiled, not run on hardware
//
#include <hip/hip_runtime.h>

// ---------------- types & helpers ----------------
typedef __attribute__((ext_vector_type(16))) __bf16 v16bf;
typedef __attribute__((ext_vector_type(8)))  float  v8f;

__device__ __forceinline__ __bf16 f2bf(float f) {
  unsigned u = __builtin_bit_cast(unsigned, f);
  unsigned r = (u + 0x7FFFu + ((u >> 16) & 1u)) >> 16;
  unsigned short s = (unsigned short)r;
  return __builtin_bit_cast(__bf16, s);
}

__device__ __forceinline__ float gelu_exact(float x) {
  return 0.5f * x * (1.0f + erff(x * 0.70710678118654752f));
}

// LDS byte-offset of a __shared__ object (generic -> addrspace(3) -> u32)
__device__ __forceinline__ unsigned lds_addr(const void* p) {
  const __attribute__((address_space(3))) void* lp =
      (const __attribute__((address_space(3))) void*)p;
  return (unsigned)(unsigned long long)lp;
}

// gfx1250 async global->LDS copy (16B per lane), tracked by ASYNCcnt
__device__ __forceinline__ void async_load_b128(unsigned lds, const void* gaddr) {
  asm volatile("global_load_async_to_lds_b128 %0, %1, off"
               :: "v"(lds), "v"(gaddr) : "memory");
}
__device__ __forceinline__ void wait_async0() {
  asm volatile("s_wait_asynccnt 0x0" ::: "memory");
}

// XOR-lane reduction helpers via ds_swizzle (stays inside the 16-lane half for m<16)
template <int MASK>
__device__ __forceinline__ float swz_xor(float x) {
  int i = __builtin_bit_cast(int, x);
  i = __builtin_amdgcn_ds_swizzle(i, (MASK << 10) | 0x1F);
  return __builtin_bit_cast(float, i);
}
__device__ __forceinline__ float red_max16(float x) {
  x = fmaxf(x, swz_xor<8>(x)); x = fmaxf(x, swz_xor<4>(x));
  x = fmaxf(x, swz_xor<2>(x)); x = fmaxf(x, swz_xor<1>(x));
  return x;
}
__device__ __forceinline__ float red_sum16(float x) {
  x += swz_xor<8>(x); x += swz_xor<4>(x); x += swz_xor<2>(x); x += swz_xor<1>(x);
  return x;
}

// fixed problem shape (from setup_inputs)
#define BB   4
#define HH   56
#define WW   56
#define NN   3136     // H*W
#define CC   256
#define HID  1024
#define PP   1376     // 144+256+400+576
#define NH   8
#define HD   32

// ---------------- depthwise 3x3 + residual (channels-last) ----------------
template <bool BF16OUT>
__global__ __launch_bounds__(256) void k_dwconv_nlc(
    const float* __restrict__ in, const float* __restrict__ w,
    const float* __restrict__ bias, void* __restrict__ outp, int Cch) {
  int n = blockIdx.x % NN;
  int b = blockIdx.x / NN;
  int hh = n / WW, ww = n % WW;
  for (int c = threadIdx.x; c < Cch; c += 256) {
    float acc = bias[c];
#pragma unroll
    for (int dh = -1; dh <= 1; dh++) {
      int h2 = hh + dh;
      if ((unsigned)h2 >= (unsigned)HH) continue;
#pragma unroll
      for (int dw = -1; dw <= 1; dw++) {
        int w2 = ww + dw;
        if ((unsigned)w2 >= (unsigned)WW) continue;
        acc += in[((size_t)(b * NN + h2 * WW + w2)) * Cch + c] *
               w[c * 9 + (dh + 1) * 3 + (dw + 1)];
      }
    }
    size_t o = ((size_t)(b * NN + n)) * Cch + c;
    float val = acc + in[o];
    if (BF16OUT) ((__bf16*)outp)[o] = f2bf(val);
    else         ((float*)outp)[o]  = val;
  }
}

// ---------------- LayerNorm over C=256 (fp32 out + optional bf16 out) -------
__global__ __launch_bounds__(256) void k_ln(
    const float* __restrict__ in, const float* __restrict__ g,
    const float* __restrict__ bt, float* __restrict__ out,
    __bf16* __restrict__ outb) {
  size_t row = blockIdx.x;
  int t = threadIdx.x;
  float v = in[row * CC + t];
  float s = v, s2 = v * v;
#pragma unroll
  for (int m = 16; m >= 1; m >>= 1) {
    s  += __shfl_xor(s,  m, 32);
    s2 += __shfl_xor(s2, m, 32);
  }
  __shared__ float ss[8], ss2[8];
  int wid = t >> 5;
  if ((t & 31) == 0) { ss[wid] = s; ss2[wid] = s2; }
  __syncthreads();
  float tot = 0.f, tot2 = 0.f;
#pragma unroll
  for (int i = 0; i < 8; i++) { tot += ss[i]; tot2 += ss2[i]; }
  float mean = tot * (1.0f / CC);
  float var  = tot2 * (1.0f / CC) - mean * mean;
  float inv  = rsqrtf(var + 1e-5f);
  float y = (v - mean) * inv * g[t] + bt[t];
  out[row * CC + t] = y;
  if (outb) outb[row * CC + t] = f2bf(y);
}

// ---------------- fp32 -> bf16 conversion (weights) ----------------
__global__ __launch_bounds__(256) void k_f2bf(
    const float* __restrict__ in, __bf16* __restrict__ out, int n) {
  int i = blockIdx.x * 256 + threadIdx.x;
  if (i < n) out[i] = f2bf(in[i]);
}

// ---------------- adaptive avg pool (all 4 scales) ----------------
__device__ __forceinline__ void pool_decode(int pidx, int& s, int& p, int& offp) {
  if (pidx < 144)      { s = 0; p = 12; offp = 0;   }
  else if (pidx < 400) { s = 1; p = 16; offp = 144; }
  else if (pidx < 800) { s = 2; p = 20; offp = 400; }
  else                 { s = 3; p = 24; offp = 800; }
}

__global__ __launch_bounds__(256) void k_pool(
    const float* __restrict__ xn, float* __restrict__ pb) {
  int pidx = blockIdx.x % PP;
  int b    = blockIdx.x / PP;
  int s, p, offp; pool_decode(pidx, s, p, offp);
  int local = pidx - offp;
  int i = local / p, j = local % p;
  int si = (i * HH) / p, ei = ((i + 1) * HH + p - 1) / p;
  int sj = (j * WW) / p, ej = ((j + 1) * WW + p - 1) / p;
  float inv = 1.0f / (float)((ei - si) * (ej - sj));
  int c = threadIdx.x;
  float acc = 0.f;
  for (int hh = si; hh < ei; hh++)
    for (int ww = sj; ww < ej; ww++)
      acc += xn[((size_t)(b * NN + hh * WW + ww)) * CC + c];
  pb[(size_t)b * (CC * PP) + (size_t)offp * CC + (size_t)c * (p * p) + local] = acc * inv;
}

__global__ __launch_bounds__(256) void k_poolconv(
    const float* __restrict__ pb,
    const float* w0, const float* b0, const float* w1, const float* b1,
    const float* w2, const float* b2, const float* w3, const float* b3,
    float* __restrict__ outc) {
  int pidx = blockIdx.x % PP;
  int b    = blockIdx.x / PP;
  int s, p, offp; pool_decode(pidx, s, p, offp);
  int local = pidx - offp;
  int i = local / p, j = local % p;
  const float* wsel = (s == 0) ? w0 : (s == 1) ? w1 : (s == 2) ? w2 : w3;
  const float* bsel = (s == 0) ? b0 : (s == 1) ? b1 : (s == 2) ? b2 : b3;
  int c = threadIdx.x;
  const float* base = pb + (size_t)b * (CC * PP) + (size_t)offp * CC + (size_t)c * (p * p);
  float center = base[i * p + j];
  float acc = bsel[c];
#pragma unroll
  for (int dh = -1; dh <= 1; dh++) {
    int ii = i + dh;
    if ((unsigned)ii >= (unsigned)p) continue;
#pragma unroll
    for (int dw = -1; dw <= 1; dw++) {
      int jj = j + dw;
      if ((unsigned)jj >= (unsigned)p) continue;
      acc += base[ii * p + jj] * wsel[c * 9 + (dh + 1) * 3 + (dw + 1)];
    }
  }
  outc[((size_t)(b * PP + pidx)) * CC + c] = center + acc;
}

// ---------------- bf16 WMMA GEMM: out = act(A @ W^T + bias) + res ----------
// A: [M,KDIM] bf16, W: [NC,KDIM] bf16. Tile 128x64, BK=32, 256 threads.
// Async global->LDS staging, double buffered; 4 WMMAs per wave per K-step.
template <int KDIM, bool BF16OUT, int ACT, bool RES>
__global__ __launch_bounds__(256) void k_gemm(
    const __bf16* __restrict__ A, const __bf16* __restrict__ W,
    const float* __restrict__ bias, const float* __restrict__ res,
    void* __restrict__ Cp, int M, int NC) {
  __shared__ __bf16 As[2][128][40];
  __shared__ __bf16 Bs[2][64][40];
  int nTilesN = NC >> 6;
  int bm = blockIdx.x / nTilesN, bn = blockIdx.x % nTilesN;
  int m0 = bm << 7, n0 = bn << 6;
  int tid = threadIdx.x, lane = tid & 31, wid = tid >> 5;
  int half = lane >> 4, l16 = lane & 15;
  int mw = wid & 3, nw = wid >> 2;   // wave = 32 rows x 32 cols
  v8f acc[2][2] = {};
  // staging assignment: 16B (8 bf16) chunk per async op
  int rA0 = tid >> 2,          cA = (tid & 3) * 8;   // A rows 0..63
  int rA1 = (tid >> 2) + 64;                          // A rows 64..127
  int rB  = tid >> 2;                                 // B rows 0..63
  const __bf16* Ab = A + (size_t)m0 * KDIM;
  const __bf16* Wb = W + (size_t)n0 * KDIM;
#define STAGE(buf, kk)                                                          \
  do {                                                                          \
    async_load_b128(lds_addr(&As[buf][rA0][cA]), Ab + (size_t)rA0 * KDIM + (kk) + cA); \
    async_load_b128(lds_addr(&As[buf][rA1][cA]), Ab + (size_t)rA1 * KDIM + (kk) + cA); \
    async_load_b128(lds_addr(&Bs[buf][rB][cA]),  Wb + (size_t)rB  * KDIM + (kk) + cA); \
  } while (0)
  STAGE(0, 0);
  wait_async0();
  __syncthreads();
  int cur = 0;
  for (int k0 = 0; k0 < KDIM; k0 += 32) {
    if (k0 + 32 < KDIM) STAGE(cur ^ 1, k0 + 32);
    v16bf a0, a1, b0, b1;
    int am = (mw << 5) + l16;
    int bnn = (nw << 5) + l16;
#pragma unroll
    for (int e = 0; e < 16; e++) {
      int ka = ((e >> 3) << 4) + half * 8 + (e & 7);
      a0[e] = As[cur][am][ka];
      a1[e] = As[cur][am + 16][ka];
    }
#pragma unroll
    for (int e = 0; e < 16; e++) {
      int kb = ((e >> 3) << 4) + half * 8 + (e & 7);
      b0[e] = Bs[cur][bnn][kb];
      b1[e] = Bs[cur][bnn + 16][kb];
    }
    acc[0][0] = __builtin_amdgcn_wmma_f32_16x16x32_bf16(false, a0, false, b0, (short)0, acc[0][0], false, false);
    acc[0][1] = __builtin_amdgcn_wmma_f32_16x16x32_bf16(false, a0, false, b1, (short)0, acc[0][1], false, false);
    acc[1][0] = __builtin_amdgcn_wmma_f32_16x16x32_bf16(false, a1, false, b0, (short)0, acc[1][0], false, false);
    acc[1][1] = __builtin_amdgcn_wmma_f32_16x16x32_bf16(false, a1, false, b1, (short)0, acc[1][1], false, false);
    wait_async0();
    __syncthreads();
    cur ^= 1;
  }
#undef STAGE
#pragma unroll
  for (int i = 0; i < 2; i++) {
#pragma unroll
    for (int j = 0; j < 2; j++) {
      int col = n0 + (nw << 5) + (j << 4) + l16;
#pragma unroll
      for (int v = 0; v < 8; v++) {
        int row = m0 + (mw << 5) + (i << 4) + v + 8 * half;
        float c = acc[i][j][v];
        if (bias) c += bias[col];
        if (ACT == 1) c = gelu_exact(c);
        if (RES) c += res[(size_t)row * NC + col];
        if (BF16OUT) ((__bf16*)Cp)[(size_t)row * NC + col] = f2bf(c);
        else         ((float*)Cp)[(size_t)row * NC + col]  = c;
      }
    }
  }
}

// ---------------- flash attention over pooled tokens ----------------
// Q: [B,N,C] bf16, KV: [B,P,2C] bf16 -> O: [B,N,C] bf16 (softmax-normalized)
__global__ __launch_bounds__(128) void k_attn(
    const __bf16* __restrict__ Q, const __bf16* __restrict__ KV,
    __bf16* __restrict__ O) {
  int blk = blockIdx.x;
  int mt = blk % 49;            // 49 tiles of 64 rows
  int h  = (blk / 49) & 7;
  int b  = blk / (49 * 8);
  int tid = threadIdx.x, lane = tid & 31, wid = tid >> 5;
  int half = lane >> 4, l16 = lane & 15;
  int n0 = mt * 64 + wid * 16;  // 16-row tile per wave
  __shared__ __bf16 Ks[2][32][40];
  __shared__ __bf16 Vs[2][32][40];
  __shared__ __bf16 Ps[4][16][40];
  // Q fragment (A layout), registers for the whole P loop
  v16bf aq;
  {
    const __bf16* qp = Q + ((size_t)(b * NN + n0 + l16)) * CC + h * HD;
#pragma unroll
    for (int e = 0; e < 16; e++) aq[e] = qp[((e >> 3) << 4) + half * 8 + (e & 7)];
  }
  float mrun[8], lrun[8];
#pragma unroll
  for (int v = 0; v < 8; v++) { mrun[v] = -1e30f; lrun[v] = 0.f; }
  v8f o0 = {}, o1 = {};
  const float scale = 0.17677669529663687f;   // 1/sqrt(32)
  int srow = tid >> 2, scol = (tid & 3) * 8;  // 128 threads -> one 16B chunk each
  const __bf16* kvb = KV + (size_t)b * PP * (2 * CC) + h * HD;
#define STAGEKV(buf, pp)                                                        \
  do {                                                                          \
    const __bf16* kp = kvb + (size_t)((pp) + srow) * (2 * CC) + scol;           \
    async_load_b128(lds_addr(&Ks[buf][srow][scol]), kp);                        \
    async_load_b128(lds_addr(&Vs[buf][srow][scol]), kp + CC);                   \
  } while (0)
  STAGEKV(0, 0);
  wait_async0();
  __syncthreads();
  int cur = 0;
  for (int p0 = 0; p0 < PP; p0 += 32) {
    if (p0 + 32 < PP) STAGEKV(cur ^ 1, p0 + 32);
    // S = Q @ K^T  (two 16x16 tiles over 32 pooled tokens)
    v16bf bk0, bk1;
#pragma unroll
    for (int e = 0; e < 16; e++) {
      int kb = ((e >> 3) << 4) + half * 8 + (e & 7);
      bk0[e] = Ks[cur][l16][kb];
      bk1[e] = Ks[cur][l16 + 16][kb];
    }
    v8f s0 = {}, s1 = {};
    s0 = __builtin_amdgcn_wmma_f32_16x16x32_bf16(false, aq, false, bk0, (short)0, s0, false, false);
    s1 = __builtin_amdgcn_wmma_f32_16x16x32_bf16(false, aq, false, bk1, (short)0, s1, false, false);
    // online softmax: rows live per-VGPR, reduce across the 16-lane half
    float p0v[8], p1v[8];
#pragma unroll
    for (int v = 0; v < 8; v++) {
      float x0 = s0[v] * scale, x1 = s1[v] * scale;
      float cm = red_max16(fmaxf(x0, x1));
      float mn = fmaxf(mrun[v], cm);
      float sc = __expf(mrun[v] - mn);
      float e0 = __expf(x0 - mn), e1 = __expf(x1 - mn);
      float rs = red_sum16(e0 + e1);
      lrun[v] = lrun[v] * sc + rs;
      mrun[v] = mn;
      o0[v] *= sc; o1[v] *= sc;
      p0v[v] = e0; p1v[v] = e1;
    }
    // C-layout -> A-layout transpose of exp-scores via LDS
#pragma unroll
    for (int v = 0; v < 8; v++) {
      int r = v + 8 * half;
      Ps[wid][r][l16]      = f2bf(p0v[v]);
      Ps[wid][r][l16 + 16] = f2bf(p1v[v]);
    }
    __syncthreads();
    v16bf apx, bv0, bv1;
#pragma unroll
    for (int e = 0; e < 16; e++) apx[e] = Ps[wid][l16][((e >> 3) << 4) + half * 8 + (e & 7)];
#pragma unroll
    for (int e = 0; e < 16; e++) {
      int kr = ((e >> 3) << 4) + half * 8 + (e & 7);
      bv0[e] = Vs[cur][kr][l16];
      bv1[e] = Vs[cur][kr][l16 + 16];
    }
    o0 = __builtin_amdgcn_wmma_f32_16x16x32_bf16(false, apx, false, bv0, (short)0, o0, false, false);
    o1 = __builtin_amdgcn_wmma_f32_16x16x32_bf16(false, apx, false, bv1, (short)0, o1, false, false);
    wait_async0();
    __syncthreads();
    cur ^= 1;
  }
#undef STAGEKV
#pragma unroll
  for (int v = 0; v < 8; v++) {
    int n = n0 + v + 8 * half;
    float inv = 1.0f / lrun[v];
    __bf16* op = O + ((size_t)(b * NN + n)) * CC + h * HD;
    op[l16]      = f2bf(o0[v] * inv);
    op[l16 + 16] = f2bf(o1[v] * inv);
  }
}

// ---------------- host orchestration ----------------
extern "C" void kernel_launch(void* const* d_in, const int* in_sizes, int n_in,
                              void* d_out, int out_size, void* d_ws, size_t ws_size,
                              hipStream_t stream) {
  (void)in_sizes; (void)n_in; (void)out_size; (void)ws_size;
  const float* x           = (const float*)d_in[0];
  const float* cpe_w       = (const float*)d_in[1];
  const float* cpe_b       = (const float*)d_in[2];
  const float* norm1_w     = (const float*)d_in[3];
  const float* norm1_b     = (const float*)d_in[4];
  const float* q_w         = (const float*)d_in[5];
  const float* kv_w        = (const float*)d_in[6];
  const float* attn_norm_w = (const float*)d_in[7];
  const float* attn_norm_b = (const float*)d_in[8];
  const float* proj_w      = (const float*)d_in[9];
  const float* proj_b      = (const float*)d_in[10];
  const float* norm2_w     = (const float*)d_in[11];
  const float* norm2_b     = (const float*)d_in[12];
  const float* fc1_w       = (const float*)d_in[13];
  const float* fc1_b       = (const float*)d_in[14];
  const float* irb_conv_w  = (const float*)d_in[15];
  const float* irb_conv_b  = (const float*)d_in[16];
  const float* fc2_w       = (const float*)d_in[17];
  const float* fc2_b       = (const float*)d_in[18];
  // d_in[19]=H, d_in[20]=W (device scalars; shapes fixed at 56x56)
  const float* dw0 = (const float*)d_in[21]; const float* db0 = (const float*)d_in[22];
  const float* dw1 = (const float*)d_in[23]; const float* db1 = (const float*)d_in[24];
  const float* dw2 = (const float*)d_in[25]; const float* db2 = (const float*)d_in[26];
  const float* dw3 = (const float*)d_in[27]; const float* db3 = (const float*)d_in[28];

  char* wsb = (char*)d_ws;
  size_t off = 0;
  auto carve = [&](size_t bytes) -> char* {
    char* p = wsb + off;
    off += (bytes + 255) & ~(size_t)255;
    return p;
  };
  const size_t MN = (size_t)BB * NN;          // 12544
  float*  x1        = (float*)carve(MN * CC * 4);
  float*  xn        = (float*)carve(MN * CC * 4);
  __bf16* xnb       = (__bf16*)carve(MN * CC * 2);
  float*  x2        = (float*)carve(MN * CC * 4);
  float*  xn2       = (float*)carve(MN * CC * 4);
  __bf16* xn2b      = (__bf16*)carve(MN * CC * 2);
  __bf16* attn_outb = (__bf16*)carve(MN * CC * 2);
  float*  hbuf      = (float*)carve(MN * HID * 4);
  __bf16* h2b       = (__bf16*)carve(MN * HID * 2);
  float*  pool_buf  = (float*)carve((size_t)BB * CC * PP * 4);
  float*  pools_cat = (float*)carve((size_t)BB * PP * CC * 4);
  float*  pools_ln  = (float*)carve((size_t)BB * PP * CC * 4);
  __bf16* pools_lnb = (__bf16*)carve((size_t)BB * PP * CC * 2);
  __bf16* qbf       = (__bf16*)carve(MN * CC * 2);
  __bf16* kvbf      = (__bf16*)carve((size_t)BB * PP * 2 * CC * 2);
  __bf16* q_wb      = (__bf16*)carve((size_t)CC * CC * 2);
  __bf16* kv_wb     = (__bf16*)carve((size_t)2 * CC * CC * 2);
  __bf16* proj_wb   = (__bf16*)carve((size_t)CC * CC * 2);
  __bf16* fc1_wb    = (__bf16*)carve((size_t)HID * CC * 2);
  __bf16* fc2_wb    = (__bf16*)carve((size_t)CC * HID * 2);

  // weight conversions to bf16 (tiny)
  k_f2bf<<<(CC * CC + 255) / 256, 256, 0, stream>>>(q_w, q_wb, CC * CC);
  k_f2bf<<<(2 * CC * CC + 255) / 256, 256, 0, stream>>>(kv_w, kv_wb, 2 * CC * CC);
  k_f2bf<<<(CC * CC + 255) / 256, 256, 0, stream>>>(proj_w, proj_wb, CC * CC);
  k_f2bf<<<(HID * CC + 255) / 256, 256, 0, stream>>>(fc1_w, fc1_wb, HID * CC);
  k_f2bf<<<(CC * HID + 255) / 256, 256, 0, stream>>>(fc2_w, fc2_wb, CC * HID);

  // 1. CPE: x1 = x + dwconv(x)
  k_dwconv_nlc<false><<<BB * NN, 256, 0, stream>>>(x, cpe_w, cpe_b, x1, CC);
  // 2. LN1 -> xn (fp32, for pooling) + xnb (bf16, GEMM operand)
  k_ln<<<(int)MN, 256, 0, stream>>>(x1, norm1_w, norm1_b, xn, xnb);
  // 3. Q projection -> bf16
  k_gemm<CC, true, 0, false><<<(12544 / 128) * (CC / 64), 256, 0, stream>>>(
      xnb, q_wb, nullptr, nullptr, qbf, 12544, CC);
  // 4. pyramid pooling + per-scale dwconv, concat channels-last
  k_pool<<<BB * PP, 256, 0, stream>>>(xn, pool_buf);
  k_poolconv<<<BB * PP, 256, 0, stream>>>(pool_buf, dw0, db0, dw1, db1, dw2, db2, dw3, db3,
                                          pools_cat);
  // 5. attn LN over pooled tokens
  k_ln<<<BB * PP, 256, 0, stream>>>(pools_cat, attn_norm_w, attn_norm_b, pools_ln, pools_lnb);
  // 6. KV projection -> bf16
  k_gemm<CC, true, 0, false><<<(5504 / 128) * (512 / 64), 256, 0, stream>>>(
      pools_lnb, kv_wb, nullptr, nullptr, kvbf, 5504, 2 * CC);
  // 7. flash attention -> bf16
  k_attn<<<BB * NH * 49, 128, 0, stream>>>(qbf, kvbf, attn_outb);
  // 8. output projection + residual(x1) -> x2
  k_gemm<CC, false, 0, true><<<(12544 / 128) * (CC / 64), 256, 0, stream>>>(
      attn_outb, proj_wb, proj_b, x1, x2, 12544, CC);
  // 9. LN2
  k_ln<<<(int)MN, 256, 0, stream>>>(x2, norm2_w, norm2_b, xn2, xn2b);
  // 10. fc1 + GELU (fp32 out: dwconv consumes it)
  k_gemm<CC, false, 1, false><<<(12544 / 128) * (HID / 64), 256, 0, stream>>>(
      xn2b, fc1_wb, fc1_b, nullptr, hbuf, 12544, HID);
  // 11. IRB dwconv: h2 = h + dwconv(h) -> bf16
  k_dwconv_nlc<true><<<BB * NN, 256, 0, stream>>>(hbuf, irb_conv_w, irb_conv_b, h2b, HID);
  // 12. fc2 + residual(x2) -> final output (fp32)
  k_gemm<HID, false, 0, true><<<(12544 / 128) * (CC / 64), 256, 0, stream>>>(
      h2b, fc2_wb, fc2_b, x2, d_out, 12544, CC);
}
